// Transformer_8881992368788
// MI455X (gfx1250) — compile-verified, matrix operand order checked by reading, not run on hardware
//
#include <hip/hip_runtime.h>
#include <hip/hip_bf16.h>

// ---------------------------------------------------------------------------
// Types for CDNA5 WMMA (wave32)
// ---------------------------------------------------------------------------
typedef __attribute__((ext_vector_type(16))) __bf16       v16bf;
typedef __attribute__((ext_vector_type(8)))  float        v8f;
typedef __attribute__((ext_vector_type(4)))  unsigned int u32x4;
typedef __attribute__((ext_vector_type(4)))  float        f32x4;
typedef __attribute__((ext_vector_type(4)))  __bf16       bf16x4;

union Frag16 { u32x4 u[2]; v16bf v; };

__device__ __forceinline__ u32x4 zero_u4() {
    u32x4 z; z[0] = 0u; z[1] = 0u; z[2] = 0u; z[3] = 0u; return z;
}

// ---------------------------------------------------------------------------
// Generic bf16 WMMA GEMM:  C = alpha * (A @ B^T) + bias  (optional ReLU)
//   A: [M,K] bf16 row-major, leading dim lda
//   B: [N,K] bf16 row-major, leading dim ldb   (second operand transposed)
//   C: [M,N] fp32 row-major, leading dim ldc
// blockIdx.z batching: z1 = z / Z2, z2 = z % Z2, per-operand element offsets
//   off = z1*s?1 + z2*s?2
// Tile: BM=128, BN=128, BK=32. 256 threads = 8 waves; wave w owns columns
// [w*16, w*16+16) and all eight 16-row M-subtiles (8 wmma per k-step).
// Double-buffered LDS (one barrier per k-step), software-pipelined global
// staging, and register ping-pong on A-fragments so the DS load of fragment
// mt+1 overlaps the WMMA of fragment mt (partial dscnt waits).
// Requires K % 32 == 0 (true for all call sites: 64/512/2048).
// ---------------------------------------------------------------------------
#define BM 128
#define BN 128
#define BK 32
#define LP 40   // LDS row pitch in bf16 elems (80B: 16B aligned, conflict-free)
#define BUFSZ (BM * LP)

__global__ __launch_bounds__(256)
void gemm_bf16_wmma(const __bf16* __restrict__ A, const __bf16* __restrict__ Bm,
                    float* __restrict__ C, const float* __restrict__ bias,
                    int M, int N, int K, int lda, int ldb, int ldc,
                    int Z2,
                    long long sA1, long long sA2,
                    long long sB1, long long sB2,
                    long long sC1, long long sC2,
                    float alpha, int relu)
{
    __shared__ __bf16 As[2 * BUFSZ];
    __shared__ __bf16 Bs[2 * BUFSZ];

    const int t    = threadIdx.x;
    const int wv   = t >> 5;        // wave id 0..7
    const int l    = t & 31;        // lane id (wave32)
    const int lrow = l & 15;        // row/col within 16x16 fragment
    const int kb   = (l >> 4) * 8;  // K phase: lanes 0-15 -> 0, 16-31 -> 8
    const int hrow = (l >> 4) * 8;  // C/D row phase

    const long long z  = blockIdx.z;
    const long long z1 = z / Z2, z2 = z % Z2;
    const __bf16* Ab = A  + z1 * sA1 + z2 * sA2;
    const __bf16* Bb = Bm + z1 * sB1 + z2 * sB2;
    float*        Cb = C  + z1 * sC1 + z2 * sC2;

    const int tile_m = blockIdx.y * BM;
    const int tile_n = blockIdx.x * BN;

    v8f acc[8];
#pragma unroll
    for (int mt = 0; mt < 8; ++mt) {
#pragma unroll
        for (int i = 0; i < 8; ++i) acc[mt][i] = 0.0f;
    }

    // Staging: 256 threads move 128 rows x 32 bf16 (two 16B chunks per thread)
    const int sr = t & 127;         // staging row
    const int sc = (t >> 7) * 16;   // staging col base (0 or 16)
    const bool aok = (tile_m + sr) < M;
    const bool bok = (tile_n + sr) < N;
    const __bf16* aRow = Ab + (long long)(tile_m + sr) * lda + sc;
    const __bf16* bRow = Bb + (long long)(tile_n + sr) * ldb + sc;

    u32x4 ra0, ra1, rb0, rb1;   // in-flight staging registers

    // ---- prologue: load tile 0, store to buffer 0 ----
    ra0 = zero_u4(); ra1 = zero_u4(); rb0 = zero_u4(); rb1 = zero_u4();
    if (aok) { ra0 = *(const u32x4*)aRow;       ra1 = *(const u32x4*)(aRow + 8); }
    if (bok) { rb0 = *(const u32x4*)bRow;       rb1 = *(const u32x4*)(bRow + 8); }
    *(u32x4*)&As[sr * LP + sc]     = ra0;
    *(u32x4*)&As[sr * LP + sc + 8] = ra1;
    *(u32x4*)&Bs[sr * LP + sc]     = rb0;
    *(u32x4*)&Bs[sr * LP + sc + 8] = rb1;
    __syncthreads();

    const int nk = K / BK;
    for (int it = 0; it < nk; ++it) {
        const int cb = (it & 1) * BUFSZ;        // compute buffer base
        const int pb = ((it + 1) & 1) * BUFSZ;  // staging buffer base

        // ---- issue global loads for tile it+1 (overlap with compute) ----
        const bool more = (it + 1) < nk;
        if (more) {
            const int k0n = (it + 1) * BK;
            ra0 = zero_u4(); ra1 = zero_u4(); rb0 = zero_u4(); rb1 = zero_u4();
            if (aok) {
                const __bf16* gp = aRow + k0n;
                ra0 = *(const u32x4*)gp;
                ra1 = *(const u32x4*)(gp + 8);
                if (it + 2 < nk)
                    __builtin_prefetch((const void*)(gp + BK), 0, 3);
            }
            if (bok) {
                const __bf16* gp = bRow + k0n;
                rb0 = *(const u32x4*)gp;
                rb1 = *(const u32x4*)(gp + 8);
                if (it + 2 < nk)
                    __builtin_prefetch((const void*)(gp + BK), 0, 3);
            }
        }

        // ---- compute from current buffer ----
        Frag16 fb;
        {
            const __bf16* bp = &Bs[cb + (wv * 16 + lrow) * LP + kb];
            fb.u[0] = *(const u32x4*)bp;
            fb.u[1] = *(const u32x4*)(bp + 16);
        }
        // Register ping-pong: load fragment mt+1 while WMMA consumes mt.
        Frag16 fa[2];
        {
            const __bf16* ap = &As[cb + lrow * LP + kb];
            fa[0].u[0] = *(const u32x4*)ap;
            fa[0].u[1] = *(const u32x4*)(ap + 16);
        }
#pragma unroll
        for (int mt = 0; mt < 8; ++mt) {
            if (mt + 1 < 8) {
                const __bf16* ap = &As[cb + ((mt + 1) * 16 + lrow) * LP + kb];
                fa[(mt + 1) & 1].u[0] = *(const u32x4*)ap;
                fa[(mt + 1) & 1].u[1] = *(const u32x4*)(ap + 16);
            }
            acc[mt] = __builtin_amdgcn_wmma_f32_16x16x32_bf16(
                false, fa[mt & 1].v, false, fb.v, (short)0, acc[mt], false, false);
        }

        // ---- commit tile it+1 to the other buffer ----
        if (more) {
            *(u32x4*)&As[pb + sr * LP + sc]     = ra0;
            *(u32x4*)&As[pb + sr * LP + sc + 8] = ra1;
            *(u32x4*)&Bs[pb + sr * LP + sc]     = rb0;
            *(u32x4*)&Bs[pb + sr * LP + sc + 8] = rb1;
            __syncthreads();
        }
    }

    // C store per 16x16 f32 layout: VGPR r -> M = hrow + r, N = lane&15
    const int gcol = tile_n + wv * 16 + lrow;
    if (gcol < N) {
        const float bv = bias ? bias[gcol] : 0.0f;
#pragma unroll
        for (int mt = 0; mt < 8; ++mt) {
            const int rbase = tile_m + mt * 16 + hrow;
#pragma unroll
            for (int r = 0; r < 8; ++r) {
                const int gr = rbase + r;
                if (gr < M) {
                    float v = acc[mt][r] * alpha + bv;
                    if (relu && v < 0.0f) v = 0.0f;
                    Cb[(long long)gr * ldc + gcol] = v;
                }
            }
        }
    }
}

// ---------------------------------------------------------------------------
// fp32 -> bf16 convert, 4 elems/thread (all sizes here are multiples of 4)
// ---------------------------------------------------------------------------
__global__ void cvt_f32_bf16_k(const float* __restrict__ s,
                               __bf16* __restrict__ d, size_t n)
{
    size_t i = ((size_t)blockIdx.x * blockDim.x + threadIdx.x) * 4;
    if (i + 3 < n) {
        f32x4 x = *(const f32x4*)(s + i);
        bf16x4 y;
        y[0] = (__bf16)x[0]; y[1] = (__bf16)x[1];
        y[2] = (__bf16)x[2]; y[3] = (__bf16)x[3];
        *(bf16x4*)(d + i) = y;
    } else {
        for (; i < n; ++i) d[i] = (__bf16)s[i];
    }
}

// ---------------------------------------------------------------------------
// V transpose-convert: vt[b,h,n,s] = v[(b*S+s)*D + h*dk + n]  (fp32 -> bf16)
// ---------------------------------------------------------------------------
__global__ void transpose_v_k(const float* __restrict__ v,
                              __bf16* __restrict__ vt,
                              int Bn, int H, int S, int dk)
{
    size_t idx = (size_t)blockIdx.x * blockDim.x + threadIdx.x;
    size_t total = (size_t)Bn * H * dk * S;
    if (idx >= total) return;
    int s = (int)(idx % S); size_t r = idx / S;
    int n = (int)(r % dk);  r /= dk;
    int h = (int)(r % H);
    int b = (int)(r / H);
    int D = H * dk;
    vt[idx] = (__bf16)v[((size_t)(b * S + s)) * D + h * dk + n];
}

// ---------------------------------------------------------------------------
// Masked softmax over last axis of scores[B*H, S, S] (in place)
// grid = (S rows, B*H), block = 256
// ---------------------------------------------------------------------------
__global__ void softmax_k(float* __restrict__ sc, const int* __restrict__ mask,
                          int H, int S, int causal)
{
    const int q = blockIdx.x;
    const int z = blockIdx.y;
    const int b = z / H;
    float* p = sc + ((size_t)z * S + q) * S;
    __shared__ float red[256];
    const int t = threadIdx.x;

    float mx = -3.4e38f;
    for (int k = t; k < S; k += 256) {
        bool ok = (mask == nullptr) || (mask[(size_t)b * S + k] != 0);
        if (causal && k > q) ok = false;
        float v = ok ? p[k] : -1.0e9f;
        p[k] = v;
        mx = fmaxf(mx, v);
    }
    red[t] = mx; __syncthreads();
    for (int s = 128; s > 0; s >>= 1) {
        if (t < s) red[t] = fmaxf(red[t], red[t + s]);
        __syncthreads();
    }
    mx = red[0]; __syncthreads();

    float sum = 0.0f;
    for (int k = t; k < S; k += 256) {
        float e = __expf(p[k] - mx);
        p[k] = e;
        sum += e;
    }
    red[t] = sum; __syncthreads();
    for (int s = 128; s > 0; s >>= 1) {
        if (t < s) red[t] += red[t + s];
        __syncthreads();
    }
    const float inv = 1.0f / red[0];
    for (int k = t; k < S; k += 256) p[k] *= inv;
}

// ---------------------------------------------------------------------------
// out = LayerNorm(a + b) * g + bt   (row length D, one row per block)
// ---------------------------------------------------------------------------
__global__ void add_ln_k(const float* __restrict__ a, const float* __restrict__ b,
                         const float* __restrict__ g, const float* __restrict__ bt,
                         float* __restrict__ out, int D)
{
    const size_t base = (size_t)blockIdx.x * D;
    __shared__ float red[256];
    const int t = threadIdx.x;

    float s = 0.0f;
    for (int i = t; i < D; i += 256) s += a[base + i] + b[base + i];
    red[t] = s; __syncthreads();
    for (int k = 128; k > 0; k >>= 1) { if (t < k) red[t] += red[t + k]; __syncthreads(); }
    const float mu = red[0] / D; __syncthreads();

    float vs = 0.0f;
    for (int i = t; i < D; i += 256) {
        float d = a[base + i] + b[base + i] - mu;
        vs += d * d;
    }
    red[t] = vs; __syncthreads();
    for (int k = 128; k > 0; k >>= 1) { if (t < k) red[t] += red[t + k]; __syncthreads(); }
    const float rstd = rsqrtf(red[0] / D + 1e-5f);

    for (int i = t; i < D; i += 256) {
        float d = a[base + i] + b[base + i] - mu;
        out[base + i] = d * rstd * g[i] + bt[i];
    }
}

// ---------------------------------------------------------------------------
// Embedding * sqrt(D) + sinusoidal positional encoding. One token per block.
// ---------------------------------------------------------------------------
__global__ void embed_k(const int* __restrict__ tok, const float* __restrict__ emb,
                        float* __restrict__ out, int S, int D, float scale)
{
    const int row = blockIdx.x;          // b*S + s
    const int s   = row % S;
    const int tk  = tok[row];
    const size_t base = (size_t)row * D;
    const float* e = emb + (size_t)tk * D;
    for (int i = threadIdx.x; i < D; i += blockDim.x) {
        int   pair = i >> 1;
        float div  = __powf(10000.0f, (2.0f * pair) / (float)D);
        float ang  = (float)s / div;
        float pe   = (i & 1) ? __cosf(ang) : __sinf(ang);
        out[base + i] = e[i] * scale + pe;
    }
}

// ---------------------------------------------------------------------------
// Host orchestration
// ---------------------------------------------------------------------------
extern "C" void kernel_launch(void* const* d_in, const int* in_sizes, int n_in,
                              void* d_out, int out_size, void* d_ws, size_t ws_size,
                              hipStream_t stream)
{
    (void)in_sizes; (void)n_in; (void)out_size; (void)ws_size;

    const int Bn = 8, S = 512, D = 512, FF = 2048, H = 8, DK = 64, L = 6, V = 32000;
    const int NTOK = Bn * S;                         // 4096
    const long long DD = (long long)D * D;           // 262144

    // --- inputs (setup_inputs dict order) ---
    const int*   src      = (const int*)  d_in[0];
    const int*   tgt      = (const int*)  d_in[1];
    const int*   src_mask = (const int*)  d_in[2];
    const int*   tgt_mask = (const int*)  d_in[3];
    const float* src_emb  = (const float*)d_in[4];
    const float* tgt_emb  = (const float*)d_in[5];
    const float* out_w    = (const float*)d_in[6];
    const float* out_b    = (const float*)d_in[7];
    const float* enc_Wq   = (const float*)d_in[8];
    const float* enc_Wk   = (const float*)d_in[9];
    const float* enc_Wv   = (const float*)d_in[10];
    const float* enc_Wo   = (const float*)d_in[11];
    const float* dec_sWq  = (const float*)d_in[12];
    const float* dec_sWk  = (const float*)d_in[13];
    const float* dec_sWv  = (const float*)d_in[14];
    const float* dec_sWo  = (const float*)d_in[15];
    const float* dec_cWq  = (const float*)d_in[16];
    const float* dec_cWk  = (const float*)d_in[17];
    const float* dec_cWv  = (const float*)d_in[18];
    const float* dec_cWo  = (const float*)d_in[19];
    const float* enc_fc1_w = (const float*)d_in[20];
    const float* enc_fc1_b = (const float*)d_in[21];
    const float* enc_fc2_w = (const float*)d_in[22];
    const float* enc_fc2_b = (const float*)d_in[23];
    const float* enc_ln1_g = (const float*)d_in[24];
    const float* enc_ln1_b = (const float*)d_in[25];
    const float* enc_ln2_g = (const float*)d_in[26];
    const float* enc_ln2_b = (const float*)d_in[27];
    const float* dec_fc1_w = (const float*)d_in[28];
    const float* dec_fc1_b = (const float*)d_in[29];
    const float* dec_fc2_w = (const float*)d_in[30];
    const float* dec_fc2_b = (const float*)d_in[31];
    const float* dec_ln1_g = (const float*)d_in[32];
    const float* dec_ln1_b = (const float*)d_in[33];
    const float* dec_ln2_g = (const float*)d_in[34];
    const float* dec_ln2_b = (const float*)d_in[35];

    // --- workspace carving ---
    char* wp = (char*)d_ws;
    auto alloc = [&](size_t bytes) -> void* {
        void* r = (void*)wp;
        wp += (bytes + 255) & ~(size_t)255;
        return r;
    };
    auto cvt = [&](const float* s, __bf16* d, size_t n) {
        size_t nt = (n + 3) / 4;
        cvt_f32_bf16_k<<<(unsigned)((nt + 255) / 256), 256, 0, stream>>>(s, d, n);
    };
    auto cvtW = [&](const float* s, size_t n) -> __bf16* {
        __bf16* d = (__bf16*)alloc(n * sizeof(__bf16));
        cvt(s, d, n);
        return d;
    };

    // bf16 weights (converted every call: deterministic, capture-safe)
    __bf16* wq_e = cvtW(enc_Wq,  (size_t)L * DD);
    __bf16* wk_e = cvtW(enc_Wk,  (size_t)L * DD);
    __bf16* wv_e = cvtW(enc_Wv,  (size_t)L * DD);
    __bf16* wo_e = cvtW(enc_Wo,  (size_t)L * DD);
    __bf16* wq_s = cvtW(dec_sWq, (size_t)L * DD);
    __bf16* wk_s = cvtW(dec_sWk, (size_t)L * DD);
    __bf16* wv_s = cvtW(dec_sWv, (size_t)L * DD);
    __bf16* wo_s = cvtW(dec_sWo, (size_t)L * DD);
    __bf16* wq_c = cvtW(dec_cWq, (size_t)L * DD);
    __bf16* wk_c = cvtW(dec_cWk, (size_t)L * DD);
    __bf16* wv_c = cvtW(dec_cWv, (size_t)L * DD);
    __bf16* wo_c = cvtW(dec_cWo, (size_t)L * DD);
    __bf16* f1_e = cvtW(enc_fc1_w, (size_t)L * FF * D);
    __bf16* f2_e = cvtW(enc_fc2_w, (size_t)L * FF * D);
    __bf16* f1_d = cvtW(dec_fc1_w, (size_t)L * FF * D);
    __bf16* f2_d = cvtW(dec_fc2_w, (size_t)L * FF * D);
    __bf16* w_ov = cvtW(out_w, (size_t)V * D);

    // activations
    float*  cur     = (float*) alloc((size_t)NTOK * D * 4);
    __bf16* curbf   = (__bf16*)alloc((size_t)NTOK * D * 2);
    float*  tmp1    = (float*) alloc((size_t)NTOK * D * 4);
    float*  tmp2    = (float*) alloc((size_t)NTOK * D * 4);
    float*  tmp3    = (float*) alloc((size_t)NTOK * D * 4);
    __bf16* tmpbf   = (__bf16*)alloc((size_t)NTOK * D * 2);
    float*  qf      = (float*) alloc((size_t)NTOK * D * 4);
    float*  kf      = (float*) alloc((size_t)NTOK * D * 4);
    float*  vf      = (float*) alloc((size_t)NTOK * D * 4);
    __bf16* qbf     = (__bf16*)alloc((size_t)NTOK * D * 2);
    __bf16* kbf     = (__bf16*)alloc((size_t)NTOK * D * 2);
    __bf16* vtbf    = (__bf16*)alloc((size_t)NTOK * D * 2);
    float*  ctxf    = (float*) alloc((size_t)NTOK * D * 4);
    __bf16* ctxbf   = (__bf16*)alloc((size_t)NTOK * D * 2);
    float*  midf    = (float*) alloc((size_t)NTOK * FF * 4);
    __bf16* midbf   = (__bf16*)alloc((size_t)NTOK * FF * 2);
    float*  fff     = (float*) alloc((size_t)NTOK * D * 4);
    float*  scores  = (float*) alloc((size_t)Bn * H * S * S * 4);
    __bf16* scorebf = (__bf16*)alloc((size_t)Bn * H * S * S * 2);
    __bf16* encbf   = (__bf16*)alloc((size_t)NTOK * D * 2);

    auto gemm = [&](const __bf16* A, const __bf16* Bw, float* Cc, const float* bias,
                    int M, int N, int K, int lda, int ldb, int ldc,
                    int Z, int Z2,
                    long long sA1, long long sA2, long long sB1, long long sB2,
                    long long sC1, long long sC2, float alpha, int relu) {
        dim3 g((N + BN - 1) / BN, (M + BM - 1) / BM, Z);
        gemm_bf16_wmma<<<g, 256, 0, stream>>>(A, Bw, Cc, bias, M, N, K,
                                              lda, ldb, ldc, Z2,
                                              sA1, sA2, sB1, sB2, sC1, sC2,
                                              alpha, relu);
    };

    // Multi-head attention: out = softmax(mask(Q K^T / sqrt(dk))) V Wo^T
    auto mha = [&](const __bf16* qin, const __bf16* kvin,
                   const __bf16* Wq, const __bf16* Wk, const __bf16* Wv,
                   const __bf16* Wo, const int* mask, int causal, float* outp) {
        gemm(qin,  Wq, qf, nullptr, NTOK, D, D, D, D, D, 1, 1, 0,0,0,0,0,0, 1.0f, 0);
        gemm(kvin, Wk, kf, nullptr, NTOK, D, D, D, D, D, 1, 1, 0,0,0,0,0,0, 1.0f, 0);
        gemm(kvin, Wv, vf, nullptr, NTOK, D, D, D, D, D, 1, 1, 0,0,0,0,0,0, 1.0f, 0);
        cvt(qf, qbf, (size_t)NTOK * D);
        cvt(kf, kbf, (size_t)NTOK * D);
        {
            size_t tot = (size_t)NTOK * D;
            transpose_v_k<<<(unsigned)((tot + 255) / 256), 256, 0, stream>>>(
                vf, vtbf, Bn, H, S, DK);
        }
        // scores[z=(b,h)] = Q_bh @ K_bh^T / 8
        gemm(qbf, kbf, scores, nullptr, S, S, DK, D, D, S, Bn * H, H,
             (long long)S * D, DK, (long long)S * D, DK,
             (long long)H * S * S, (long long)S * S, 0.125f, 0);
        softmax_k<<<dim3(S, Bn * H), 256, 0, stream>>>(scores, mask, H, S, causal);
        cvt(scores, scorebf, (size_t)Bn * H * S * S);
        // ctx[b,s,h*dk+n] = attn_bh @ V_bh   (B operand = V^T per head)
        gemm(scorebf, vtbf, ctxf, nullptr, S, DK, S, S, S, D, Bn * H, H,
             (long long)H * S * S, (long long)S * S,
             (long long)H * DK * S, (long long)DK * S,
             (long long)S * D, DK, 1.0f, 0);
        cvt(ctxf, ctxbf, (size_t)NTOK * D);
        gemm(ctxbf, Wo, outp, nullptr, NTOK, D, D, D, D, D, 1, 1, 0,0,0,0,0,0, 1.0f, 0);
    };

    auto ffn = [&](const float* xin, const __bf16* fc1w, const float* fc1b,
                   const __bf16* fc2w, const float* fc2b, float* outp) {
        cvt(xin, tmpbf, (size_t)NTOK * D);
        gemm(tmpbf, fc1w, midf, fc1b, NTOK, FF, D, D, D, FF, 1, 1,
             0,0,0,0,0,0, 1.0f, 1);
        cvt(midf, midbf, (size_t)NTOK * FF);
        gemm(midbf, fc2w, outp, fc2b, NTOK, D, FF, FF, FF, D, 1, 1,
             0,0,0,0,0,0, 1.0f, 0);
    };

    const float scale = 22.627416997969522f;  // sqrt(512)

    // ----------------------------- Encoder -----------------------------
    embed_k<<<NTOK, 256, 0, stream>>>(src, src_emb, cur, S, D, scale);
    for (int i = 0; i < L; ++i) {
        cvt(cur, curbf, (size_t)NTOK * D);
        mha(curbf, curbf, wq_e + i * DD, wk_e + i * DD, wv_e + i * DD,
            wo_e + i * DD, src_mask, 0, tmp1);
        add_ln_k<<<NTOK, 256, 0, stream>>>(tmp1, cur, enc_ln1_g + (size_t)i * D,
                                           enc_ln1_b + (size_t)i * D, tmp2, D);
        ffn(tmp2, f1_e + (size_t)i * FF * D, enc_fc1_b + (size_t)i * FF,
            f2_e + (size_t)i * FF * D, enc_fc2_b + (size_t)i * D, fff);
        add_ln_k<<<NTOK, 256, 0, stream>>>(fff, tmp2, enc_ln2_g + (size_t)i * D,
                                           enc_ln2_b + (size_t)i * D, cur, D);
    }
    cvt(cur, encbf, (size_t)NTOK * D);   // encoder output (bf16 for cross-attn)

    // ----------------------------- Decoder -----------------------------
    embed_k<<<NTOK, 256, 0, stream>>>(tgt, tgt_emb, cur, S, D, scale);
    for (int i = 0; i < L; ++i) {
        cvt(cur, curbf, (size_t)NTOK * D);
        // masked self-attention
        mha(curbf, curbf, wq_s + i * DD, wk_s + i * DD, wv_s + i * DD,
            wo_s + i * DD, tgt_mask, 1, tmp1);
        add_ln_k<<<NTOK, 256, 0, stream>>>(tmp1, cur, dec_ln1_g + (size_t)i * D,
                                           dec_ln1_b + (size_t)i * D, tmp2, D); // out1
        // cross-attention: query = raw layer input y (faithful quirk)
        mha(curbf, encbf, wq_c + i * DD, wk_c + i * DD, wv_c + i * DD,
            wo_c + i * DD, src_mask, 0, tmp1);
        add_ln_k<<<NTOK, 256, 0, stream>>>(tmp1, tmp2, dec_ln1_g + (size_t)i * D,
                                           dec_ln1_b + (size_t)i * D, tmp3, D); // out2
        ffn(tmp3, f1_d + (size_t)i * FF * D, dec_fc1_b + (size_t)i * FF,
            f2_d + (size_t)i * FF * D, dec_fc2_b + (size_t)i * D, fff);
        add_ln_k<<<NTOK, 256, 0, stream>>>(fff, tmp3, dec_ln2_g + (size_t)i * D,
                                           dec_ln2_b + (size_t)i * D, cur, D);
    }

    // ------------------------ Output projection -------------------------
    cvt(cur, curbf, (size_t)NTOK * D);
    gemm(curbf, w_ov, (float*)d_out, out_b, NTOK, V, D, D, D, V, 1, 1,
         0,0,0,0,0,0, 1.0f, 0);
}
